// DogeAttention_56349970923543
// MI455X (gfx1250) — matrix operand order repaired and checked by reading, not compiled, not run on hardware
//
#include <hip/hip_runtime.h>
#include <hip/hip_bf16.h>
#include <cfloat>
#include <math.h>

#define S_LEN 3072
#define D_MODEL 1024
#define NH 8
#define NKV 4
#define HDIM 128
#define KEEPK 2048
#define EPSF 1e-6f
#define SCALEF 0.08838834764831843f  // 1/sqrt(128)

typedef __bf16 bf16;
typedef __attribute__((ext_vector_type(16))) __bf16 v16bf;
typedef __attribute__((ext_vector_type(8)))  float  v8f;

// ---------------- WMMA fragment loaders (wave32, 16x16x32 bf16) ----------------
// A-matrix 16x32 (MxK): lane L: m = L%16; K-half offset = 16*(L/16);
// VGPR j pair covers K = 2j (+half offset), +1
__device__ __forceinline__ v16bf load_a_frag(const bf16* A, int lda, int m0, int k0, int lane) {
  int m  = m0 + (lane & 15);
  int kh = (lane >> 4) << 4;
  const bf16* row = A + (size_t)m * lda + k0 + kh;
  v16bf a;
#pragma unroll
  for (int j = 0; j < 8; ++j) {
    int kb = j << 1;
    a[2 * j]     = row[kb];
    a[2 * j + 1] = row[kb + 1];
  }
  return a;
}

// B-matrix 32x16 (KxN) where source is row-major [N,K] (e.g. weights W[n][k]):
// lane L: n = L%16; K-half offset = 16*(L/16); VGPR j covers K = 2j,2j+1 (contiguous in W row)
__device__ __forceinline__ v16bf load_b_frag_nk(const bf16* W, int ldw, int n0, int k0, int lane) {
  int n    = n0 + (lane & 15);
  int koff = (lane >> 4) << 4;
  const bf16* row = W + (size_t)n * ldw + k0 + koff;
  v16bf b;
#pragma unroll
  for (int j = 0; j < 8; ++j) {
    b[2 * j]     = row[2 * j];
    b[2 * j + 1] = row[2 * j + 1];
  }
  return b;
}

// B-matrix 32x16 (KxN) where source is row-major [K, ld] (e.g. V[k][d]); rows clamped to kmax
__device__ __forceinline__ v16bf load_b_frag_kn(const bf16* V, int ld, int k0, int n0, int lane, int kmax) {
  int n    = n0 + (lane & 15);
  int koff = k0 + ((lane >> 4) << 4);
  v16bf b;
#pragma unroll
  for (int j = 0; j < 8; ++j) {
    int r0 = koff + 2 * j;     if (r0 > kmax) r0 = kmax;
    int r1 = koff + 2 * j + 1; if (r1 > kmax) r1 = kmax;
    b[2 * j]     = V[(size_t)r0 * ld + n];
    b[2 * j + 1] = V[(size_t)r1 * ld + n];
  }
  return b;
}

// ---------------- fp32 -> bf16 convert ----------------
__global__ void cvt_bf16_kernel(const float* __restrict__ src, bf16* __restrict__ dst, int n) {
  int i = blockIdx.x * blockDim.x + threadIdx.x;
  if (i < n) dst[i] = (bf16)src[i];
}

// ------ GEMM: C[M,N] = A[M,K] * B[N,K]^T (bf16 in, f32 out), 64x64 per wave -------
// 4x4 register blocking: per 32-deep k-step, 8 fragment loads feed 16 WMMAs
// (32 FLOP/byte from L2 instead of 8 for the naive 1-tile version).
__global__ __launch_bounds__(256) void gemm_bf16_wmma(const bf16* __restrict__ A,
                                                      const bf16* __restrict__ B,
                                                      float* __restrict__ C,
                                                      int M, int N, int K) {
  int wave = blockIdx.x * (blockDim.x >> 5) + (threadIdx.x >> 5);
  int lane = threadIdx.x & 31;
  int nblk = N >> 6;
  int m0 = (wave / nblk) << 6;
  int n0 = (wave % nblk) << 6;

  v8f acc[4][4];
#pragma unroll
  for (int i = 0; i < 4; ++i)
#pragma unroll
    for (int j = 0; j < 4; ++j) acc[i][j] = (v8f){};

  for (int k0 = 0; k0 < K; k0 += 32) {
    v16bf af[4], bfr[4];
#pragma unroll
    for (int i = 0; i < 4; ++i) af[i] = load_a_frag(A, K, m0 + 16 * i, k0, lane);
#pragma unroll
    for (int j = 0; j < 4; ++j) bfr[j] = load_b_frag_nk(B, K, n0 + 16 * j, k0, lane);
    // prefetch next k-chunk, one cacheline per lane across the 64 rows of A/B
    __builtin_prefetch(A + (size_t)(m0 + lane * 2) * K + k0 + 32);
    __builtin_prefetch(B + (size_t)(n0 + lane * 2) * K + k0 + 32);
#pragma unroll
    for (int i = 0; i < 4; ++i)
#pragma unroll
      for (int j = 0; j < 4; ++j)
        acc[i][j] = __builtin_amdgcn_wmma_f32_16x16x32_bf16(false, af[i], false, bfr[j],
                                                            (short)0, acc[i][j], false, false);
  }

  int nl = lane & 15;
  int mb = (lane >> 4) << 3;
#pragma unroll
  for (int i = 0; i < 4; ++i)
#pragma unroll
    for (int j = 0; j < 4; ++j) {
      int n = n0 + 16 * j + nl;
      int m = m0 + 16 * i + mb;
#pragma unroll
      for (int r = 0; r < 8; ++r) C[(size_t)(m + r) * N + n] = acc[i][j][r];
    }
}

// ---------------- dt_states[h][s] = exp(A[h] * softplus(v_flat[s] . Wdt[h])) --------
__global__ void dt_mask_kernel(const float* __restrict__ Vf, const float* __restrict__ Wdt,
                               const float* __restrict__ Av, float* __restrict__ dts) {
  int i = blockIdx.x * blockDim.x + threadIdx.x;
  if (i >= S_LEN * NH) return;
  int s = i / NH, h = i % NH;
  const float* v = Vf + (size_t)s * (NKV * HDIM);
  const float* w = Wdt + (size_t)h * (NKV * HDIM);
  float acc = 0.f;
  for (int j = 0; j < NKV * HDIM; ++j) acc = fmaf(v[j], w[j], acc);
  float sp = (acc > 20.f) ? acc : log1pf(expf(acc));
  dts[(size_t)h * S_LEN + s] = expf(Av[h] * sp);
}

// ---------------- RMS-norm + RoPE, write bf16 [nh, S, HDIM] ----------------
__global__ __launch_bounds__(HDIM) void norm_rope_kernel(const float* __restrict__ X,
                                                         const float* __restrict__ w,
                                                         const float* __restrict__ cs,
                                                         const float* __restrict__ sn,
                                                         bf16* __restrict__ out, int nh) {
  int h = blockIdx.x % nh, s = blockIdx.x / nh, d = threadIdx.x;
  float x = X[((size_t)s * nh + h) * HDIM + d];
  __shared__ float red[HDIM];
  red[d] = x * x;
  __syncthreads();
  for (int offc = HDIM / 2; offc > 0; offc >>= 1) {
    if (d < offc) red[d] += red[d + offc];
    __syncthreads();
  }
  float xn = x * rsqrtf(red[0] * (1.0f / HDIM) + EPSF) * w[d];
  __syncthreads();
  red[d] = xn;
  __syncthreads();
  float rot = (d < HDIM / 2) ? -red[d + HDIM / 2] : red[d - HDIM / 2];
  float val = xn * cs[(size_t)s * HDIM + d] + rot * sn[(size_t)s * HDIM + d];
  out[((size_t)h * S_LEN + s) * HDIM + d] = (bf16)val;
}

// ---------------- V f32 [S, NKV*HDIM] -> bf16 [NKV, S, HDIM] ----------------
__global__ void v_to_bf16_kernel(const float* __restrict__ Vf, bf16* __restrict__ out) {
  int i = blockIdx.x * blockDim.x + threadIdx.x;
  if (i >= S_LEN * NKV * HDIM) return;
  int d = i % HDIM;
  int t = i / HDIM;
  int hk = t % NKV, s = t / NKV;
  out[((size_t)hk * S_LEN + s) * HDIM + d] = (bf16)Vf[i];
}

// ---------------- per-(h,q) top-K threshold via bit-level binary search ------------
// dt values are strictly positive => uint bit order == float order.
__global__ void thresh_kernel(const float* __restrict__ dts, float* __restrict__ thr) {
  int i = blockIdx.x * blockDim.x + threadIdx.x;
  if (i >= NH * S_LEN) return;
  int h = i / S_LEN, q = i % S_LEN;
  if (q + 1 <= KEEPK) { thr[i] = 0.f; return; }
  const float* row = dts + (size_t)h * S_LEN;
  unsigned lo = 0u, hi = 0x7F7FFFFFu;
  // largest t with count(row[k<=q] >= t) >= KEEPK
  for (int it = 0; it < 31; ++it) {
    if (lo >= hi) break;
    unsigned mid = lo + ((hi - lo + 1) >> 1);
    int cnt = 0;
    for (int k = 0; k <= q; ++k) cnt += (__float_as_uint(row[k]) >= mid) ? 1 : 0;
    if (cnt >= KEEPK) lo = mid; else hi = mid - 1;
  }
  thr[i] = __uint_as_float(lo);
}

// ---------------- flash attention: one wave per (head, 16-query tile) --------------
__global__ __launch_bounds__(128) void flash_attn_kernel(const bf16* __restrict__ Qb,
                                                         const bf16* __restrict__ Kb,
                                                         const bf16* __restrict__ Vb,
                                                         const float* __restrict__ dts,
                                                         const float* __restrict__ thr,
                                                         bf16* __restrict__ Ob) {
  __shared__ bf16 pbuf[4][16 * 32];  // per-wave P tile (16 q x 32 k), bf16
  int wid = threadIdx.x >> 5;
  int lane = threadIdx.x & 31;
  int gw = blockIdx.x * 4 + wid;
  int qtiles = S_LEN / 16;
  int h  = gw / qtiles;
  int q0 = (gw % qtiles) << 4;
  int hk = h / (NH / NKV);
  const bf16* Qh = Qb + ((size_t)h * S_LEN + q0) * HDIM;
  const bf16* Kh = Kb + (size_t)hk * S_LEN * HDIM;
  const bf16* Vh = Vb + (size_t)hk * S_LEN * HDIM;
  const float* dth = dts + (size_t)h * S_LEN;

  int hf = lane >> 4;     // half (0/1)
  int nl = lane & 15;

  // Q fragments: 16 x 128, four K-chunks of 32
  v16bf qf[4];
#pragma unroll
  for (int c = 0; c < 4; ++c) qf[c] = load_a_frag(Qh, HDIM, 0, c * 32, lane);

  v8f o[8];
#pragma unroll
  for (int t = 0; t < 8; ++t) o[t] = (v8f){};
  float rmax[8], rsum[8], thrr[8];
#pragma unroll
  for (int r = 0; r < 8; ++r) {
    rmax[r] = -FLT_MAX;
    rsum[r] = 0.f;
    thrr[r] = thr[(size_t)h * S_LEN + q0 + r + 8 * hf];
  }

  int kend = q0 + 16;  // exclusive causal bound for this tile
  for (int k0 = 0; k0 < kend; k0 += 32) {
    float sc[2][8];
#pragma unroll
    for (int sub = 0; sub < 2; ++sub) {
      int kk = k0 + sub * 16;
      if (kk < kend) {  // wave-uniform
        v8f acc = {};
#pragma unroll
        for (int c = 0; c < 4; ++c) {
          v16bf kfr = load_b_frag_nk(Kh, HDIM, kk, c * 32, lane);
          acc = __builtin_amdgcn_wmma_f32_16x16x32_bf16(false, qf[c], false, kfr, (short)0, acc, false, false);
        }
        int key = kk + nl;
        float dt_here = dth[key];
#pragma unroll
        for (int r = 0; r < 8; ++r) {
          int qrow = q0 + r + 8 * hf;
          bool keep = (key <= qrow) && (dt_here >= thrr[r]);
          sc[sub][r] = keep ? (acc[r] * SCALEF + dt_here) : -FLT_MAX;
        }
      } else {
#pragma unroll
        for (int r = 0; r < 8; ++r) sc[sub][r] = -FLT_MAX;
      }
    }
    // online softmax per row r (row values live across the 16 lanes of each half)
#pragma unroll
    for (int r = 0; r < 8; ++r) {
      float bm = fmaxf(sc[0][r], sc[1][r]);
#pragma unroll
      for (int m = 1; m < 16; m <<= 1) bm = fmaxf(bm, __shfl_xor(bm, m, 32));
      float nmax = fmaxf(rmax[r], bm);
      float alpha = expf(rmax[r] - nmax);
      float p0 = expf(sc[0][r] - nmax);
      float p1 = expf(sc[1][r] - nmax);
      float ps = p0 + p1;
#pragma unroll
      for (int m = 1; m < 16; m <<= 1) ps += __shfl_xor(ps, m, 32);
      rsum[r] = rsum[r] * alpha + ps;
      rmax[r] = nmax;
#pragma unroll
      for (int t = 0; t < 8; ++t) o[t][r] *= alpha;
      // stash P into LDS as bf16: row (r + 8*hf), cols sub*16 + nl
      pbuf[wid][(r + 8 * hf) * 32 + 0 * 16 + nl] = (bf16)p0;
      pbuf[wid][(r + 8 * hf) * 32 + 1 * 16 + nl] = (bf16)p1;
    }
    // wave-local LDS ordering before cross-lane re-fragmentation
    asm volatile("s_wait_dscnt 0x0" ::: "memory");
    v16bf pa = load_a_frag((const bf16*)&pbuf[wid][0], 32, 0, 0, lane);
    // O(16x128) += P(16x32) * V(32x128)
#pragma unroll
    for (int t = 0; t < 8; ++t) {
      v16bf vb = load_b_frag_kn(Vh, HDIM, k0, t * 16, lane, S_LEN - 1);
      o[t] = __builtin_amdgcn_wmma_f32_16x16x32_bf16(false, pa, false, vb, (short)0, o[t], false, false);
    }
    asm volatile("s_wait_dscnt 0x0" ::: "memory");
  }

  // normalize and write bf16 output [S, NH*HDIM]
#pragma unroll
  for (int t = 0; t < 8; ++t) {
    int d = t * 16 + nl;
#pragma unroll
    for (int r = 0; r < 8; ++r) {
      int qrow = q0 + r + 8 * hf;
      float val = o[t][r] / rsum[r];
      Ob[(size_t)qrow * (NH * HDIM) + h * HDIM + d] = (bf16)val;
    }
  }
}

// ------------------------------- host launcher -------------------------------
extern "C" void kernel_launch(void* const* d_in, const int* in_sizes, int n_in,
                              void* d_out, int out_size, void* d_ws, size_t ws_size,
                              hipStream_t stream) {
  const float* hs   = (const float*)d_in[0];
  const float* cosT = (const float*)d_in[1];
  const float* sinT = (const float*)d_in[2];
  // d_in[3] = attention_mask (causal) -- derived analytically, unused
  const float* Wq   = (const float*)d_in[4];
  const float* Wk   = (const float*)d_in[5];
  const float* Wv   = (const float*)d_in[6];
  const float* Wdt  = (const float*)d_in[7];
  const float* Av   = (const float*)d_in[8];
  const float* qnw  = (const float*)d_in[9];
  const float* knw  = (const float*)d_in[10];
  const float* Wo   = (const float*)d_in[11];
  float* out = (float*)d_out;

  char* ws = (char*)d_ws;
  size_t off = 0;
  auto alloc = [&](size_t bytes) -> void* {
    void* p = ws + off;
    off += (bytes + 255) & ~(size_t)255;
    return p;
  };

  bf16* Xb   = (bf16*)alloc((size_t)S_LEN * D_MODEL * 2);
  bf16* Wqb  = (bf16*)alloc((size_t)D_MODEL * D_MODEL * 2);
  bf16* Wkb  = (bf16*)alloc((size_t)(NKV * HDIM) * D_MODEL * 2);
  bf16* Wvb  = (bf16*)alloc((size_t)(NKV * HDIM) * D_MODEL * 2);
  bf16* Wob  = (bf16*)alloc((size_t)D_MODEL * D_MODEL * 2);
  float* Qf  = (float*)alloc((size_t)S_LEN * D_MODEL * 4);
  float* Kf  = (float*)alloc((size_t)S_LEN * NKV * HDIM * 4);
  float* Vf  = (float*)alloc((size_t)S_LEN * NKV * HDIM * 4);
  float* dts = (float*)alloc((size_t)NH * S_LEN * 4);
  float* thr = (float*)alloc((size_t)NH * S_LEN * 4);
  bf16* Qb   = (bf16*)alloc((size_t)NH * S_LEN * HDIM * 2);
  bf16* Kb   = (bf16*)alloc((size_t)NKV * S_LEN * HDIM * 2);
  bf16* Vb   = (bf16*)alloc((size_t)NKV * S_LEN * HDIM * 2);
  bf16* Ob   = (bf16*)alloc((size_t)S_LEN * D_MODEL * 2);

  const int CB = 256;
  // 1) converts
  cvt_bf16_kernel<<<(S_LEN * D_MODEL + CB - 1) / CB, CB, 0, stream>>>(hs, Xb, S_LEN * D_MODEL);
  cvt_bf16_kernel<<<(D_MODEL * D_MODEL + CB - 1) / CB, CB, 0, stream>>>(Wq, Wqb, D_MODEL * D_MODEL);
  cvt_bf16_kernel<<<(NKV * HDIM * D_MODEL + CB - 1) / CB, CB, 0, stream>>>(Wk, Wkb, NKV * HDIM * D_MODEL);
  cvt_bf16_kernel<<<(NKV * HDIM * D_MODEL + CB - 1) / CB, CB, 0, stream>>>(Wv, Wvb, NKV * HDIM * D_MODEL);
  cvt_bf16_kernel<<<(D_MODEL * D_MODEL + CB - 1) / CB, CB, 0, stream>>>(Wo, Wob, D_MODEL * D_MODEL);

  // 2) QKV projections via WMMA GEMM (C = A * B^T), 64x64 per wave
  {
    int waves = (S_LEN / 64) * (D_MODEL / 64);  // 768
    gemm_bf16_wmma<<<waves / 8, 256, 0, stream>>>(Xb, Wqb, Qf, S_LEN, D_MODEL, D_MODEL);
  }
  {
    int waves = (S_LEN / 64) * ((NKV * HDIM) / 64);  // 384
    gemm_bf16_wmma<<<waves / 8, 256, 0, stream>>>(Xb, Wkb, Kf, S_LEN, NKV * HDIM, D_MODEL);
    gemm_bf16_wmma<<<waves / 8, 256, 0, stream>>>(Xb, Wvb, Vf, S_LEN, NKV * HDIM, D_MODEL);
  }

  // 3) dynamic mask values + thresholds
  dt_mask_kernel<<<(S_LEN * NH + CB - 1) / CB, CB, 0, stream>>>(Vf, Wdt, Av, dts);
  thresh_kernel<<<(NH * S_LEN + CB - 1) / CB, CB, 0, stream>>>(dts, thr);

  // 4) RMS-norm + RoPE -> bf16 [h, S, HDIM]
  norm_rope_kernel<<<S_LEN * NH, HDIM, 0, stream>>>(Qf, qnw, cosT, sinT, Qb, NH);
  norm_rope_kernel<<<S_LEN * NKV, HDIM, 0, stream>>>(Kf, knw, cosT, sinT, Kb, NKV);
  v_to_bf16_kernel<<<(S_LEN * NKV * HDIM + CB - 1) / CB, CB, 0, stream>>>(Vf, Vb);

  // 5) flash attention (WMMA scores + WMMA P*V)
  {
    int waves = NH * (S_LEN / 16);  // 1536
    flash_attn_kernel<<<waves / 4, 128, 0, stream>>>(Qb, Kb, Vb, dts, thr, Ob);
  }

  // 6) output projection
  {
    int waves = (S_LEN / 64) * (D_MODEL / 64);  // 768
    gemm_bf16_wmma<<<waves / 8, 256, 0, stream>>>(Ob, Wob, out, S_LEN, D_MODEL, D_MODEL);
  }
}
